// DeformableDynamicKernel2D_27736898797748
// MI455X (gfx1250) — matrix operand order repaired and checked by reading, not compiled, not run on hardware
//
#include <hip/hip_runtime.h>
#include <hip/hip_bf16.h>
#include <math.h>

typedef _Float16 f16;
typedef __attribute__((ext_vector_type(8)))  _Float16 v8h;
typedef __attribute__((ext_vector_type(16))) _Float16 v16h;
typedef __attribute__((ext_vector_type(8)))  float    v8f;

// Problem constants (match reference)
static constexpr int B_DIM  = 2;
static constexpr int C_DIM  = 64;
static constexpr int H_DIM  = 256;
static constexpr int W_DIM  = 256;
static constexpr int N_PTS  = 32768;
static constexpr int K_TAPS = 9;

static constexpr int PTS     = 16;   // points per block (one WMMA M-tile)
static constexpr int THREADS = 128;  // 4 x wave32

// ---- WMMA fragment loaders (wave32, contiguous k order k = 16*(lane/16) + e in A and B) ----------

// A matrix 16x32 f16, from row-major [16][64] LDS tile.
// lane m=l&15; lanes<16 hold K {kb+0..15}, lanes>=16 hold
// K {kb+16..31}.  Each half is a contiguous, 16B-aligned 8-half run.
__device__ __forceinline__ v16h load_A16(const f16* s, int kbase) {
  const int l   = threadIdx.x & 31;
  const int m   = l & 15;
  const int khi = (l >> 4) << 4;  // 0 or 16
  const f16* row = s + m * 64 + kbase + khi;
  v8h lo = *(const v8h*)(row);        // K = kbase+khi .. +7      (ds_load_b128)
  v8h hi = *(const v8h*)(row + 8);    // K = kbase+khi+8 .. +15   (ds_load_b128)
  return __builtin_shufflevector(lo, hi, 0, 1, 2, 3, 4, 5, 6, 7,
                                         8, 9, 10, 11, 12, 13, 14, 15);
}

// B matrix fragment from pre-swizzled LDS: frag-major [frag][lane][16 halves].
// One contiguous 32-byte per-lane read -> 2 x ds_load_b128, zero packing VALU.
__device__ __forceinline__ v16h load_Bsw(const f16* s, int frag) {
  const int l = threadIdx.x & 31;
  return *(const v16h*)(s + (frag * 32 + l) * 16);
}

// Bilinear setup identical to reference clamping semantics.
__device__ __forceinline__ void samp_setup(float gx, float gy,
                                           int& x0, int& y0, int& x1, int& y1,
                                           float& wx, float& wy) {
  float ix = (gx + 1.0f) * 0.5f * (float)(W_DIM - 1);
  float iy = (gy + 1.0f) * 0.5f * (float)(H_DIM - 1);
  ix = fminf(fmaxf(ix, 0.0f), (float)(W_DIM - 1));
  iy = fminf(fmaxf(iy, 0.0f), (float)(H_DIM - 1));
  float fx = floorf(ix), fy = floorf(iy);
  wx = ix - fx; wy = iy - fy;
  x0 = (int)fx; y0 = (int)fy;
  x1 = min(x0 + 1, W_DIM - 1);
  y1 = min(y0 + 1, H_DIM - 1);
}

__global__ __launch_bounds__(THREADS)
void deform_dyn_fused_kernel(const float* __restrict__ feat,
                             const float* __restrict__ coords,
                             const float* __restrict__ W1, const float* __restrict__ b1,
                             const float* __restrict__ Wr, const float* __restrict__ br,
                             const float* __restrict__ W2, const float* __restrict__ b2,
                             float* __restrict__ out) {
  // ---- LDS staging (~32 KB) ----
  // Weight matrices pre-swizzled to WMMA B-fragment layout:
  //   frag = ntile*2 + kchunk ; element (frag, lane, e) =
  //   W[kchunk*32 + ((lane>>4)<<4) + e][ntile*16 + (lane&15)]
  __shared__ alignas(32) f16   sW1sw[4 * 2 * 32 * 16];   // 8 KB
  __shared__ alignas(32) f16   sWrsw[4 * 2 * 32 * 16];   // 8 KB
  __shared__ alignas(32) f16   sW2sw[2 * 2 * 32 * 16];   // 4 KB (27 cols pad 32)
  __shared__ float sW1t[2 * 64];                          // W1 coord rows (f32)
  __shared__ float sB1[64];
  __shared__ float sBr[64];
  __shared__ float sB2[32];
  __shared__ alignas(32) f16   sAf[PTS * 64];             // anchor features (A)
  __shared__ alignas(32) f16   sH1[PTS * 64];             // hidden after layer 1
  __shared__ alignas(32) f16   sH2[PTS * 64];             // hidden after residual
  __shared__ float sRout[PTS * 32];                       // routing out (pad 32)
  __shared__ float sCx[PTS], sCy[PTS];
  __shared__ int   sFB[PTS];                              // per-point batch base
  __shared__ int   sAx0[PTS], sAy0[PTS], sAx1[PTS], sAy1[PTS];
  __shared__ float sAwx[PTS], sAwy[PTS];
  __shared__ int   sSx0[PTS * K_TAPS], sSy0[PTS * K_TAPS];
  __shared__ int   sSx1[PTS * K_TAPS], sSy1[PTS * K_TAPS];
  __shared__ float sSwx[PTS * K_TAPS], sSwy[PTS * K_TAPS], sSwk[PTS * K_TAPS];

  const int tid  = threadIdx.x;
  const int gp0  = blockIdx.x * PTS;
  const int wave = tid >> 5;
  const int lane = tid & 31;
  const int ln   = lane & 15;
  const int mhi  = (lane >> 4) << 3;   // C/D row offset: 0 or 8

  // Warm L2 with weight lines (gfx1250 global_prefetch path).
  __builtin_prefetch(W1, 0, 1);
  __builtin_prefetch(Wr, 0, 1);
  __builtin_prefetch(W2, 0, 1);

  // ---- Phase 0: per-point anchor setup ----
  if (tid < PTS) {
    const int gp = gp0 + tid;
    const int b  = gp / N_PTS;
    const float cx = coords[2 * gp + 0];
    const float cy = coords[2 * gp + 1];
    sCx[tid] = cx; sCy[tid] = cy;
    sFB[tid] = b * (C_DIM * H_DIM * W_DIM);
    int x0, y0, x1, y1; float wx, wy;
    samp_setup(cx, cy, x0, y0, x1, y1, wx, wy);
    sAx0[tid] = x0; sAy0[tid] = y0; sAx1[tid] = x1; sAy1[tid] = y1;
    sAwx[tid] = wx; sAwy[tid] = wy;
  }

  // ---- Phase 0b: stage weights into LDS, swizzled to B-fragment layout ----
  // Iterate the SOURCE linearly so global reads stay coalesced; scatter on
  // cheap LDS b16 stores.
  for (int j = tid; j < 64 * 64; j += THREADS) {
    const int k = j >> 6, n = j & 63;
    const int t = n >> 4, kb = k >> 5, kk = k & 31;
    const int ll = ((kk >> 4) << 4) | (n & 15);
    const int e  = kk & 15;
    const int dst = ((t * 2 + kb) * 32 + ll) * 16 + e;
    sW1sw[dst] = (f16)W1[j];
    sWrsw[dst] = (f16)Wr[j];
  }
  for (int i = tid; i < 2 * 2 * 32 * 16; i += THREADS) {  // W2: dest-linear
    const int e = i & 15, ll = (i >> 4) & 31, frag = i >> 9;
    const int t = frag >> 1, kb = frag & 1;
    const int k = kb * 32 + ((ll >> 4) << 4) + e;
    const int n = 16 * t + (ll & 15);
    sW2sw[i] = (n < 27) ? (f16)W2[k * 27 + n] : (f16)0.0f;
  }
  for (int i = tid; i < 2 * 64; i += THREADS) sW1t[i] = W1[64 * 64 + i];
  for (int i = tid; i < 64; i += THREADS) { sB1[i] = b1[i]; sBr[i] = br[i]; }
  for (int i = tid; i < 32; i += THREADS) sB2[i] = (i < 27) ? b2[i] : 0.0f;
  __syncthreads();

  // ---- Phase 1: anchor bilinear gather -> A matrix (16 pts x 64 ch), f16 ----
#pragma unroll
  for (int i = 0; i < 8; ++i) {
    const int idx = tid * 8 + i;
    const int p = idx >> 6, c = idx & 63;
    const float* fb = feat + (size_t)sFB[p] + (size_t)c * (H_DIM * W_DIM);
    const float wx = sAwx[p], wy = sAwy[p];
    const float v00 = fb[sAy0[p] * W_DIM + sAx0[p]];
    const float v01 = fb[sAy0[p] * W_DIM + sAx1[p]];
    const float v10 = fb[sAy1[p] * W_DIM + sAx0[p]];
    const float v11 = fb[sAy1[p] * W_DIM + sAx1[p]];
    const float v = v00 * (1.0f - wx) * (1.0f - wy) + v01 * wx * (1.0f - wy)
                  + v10 * (1.0f - wx) * wy          + v11 * wx * wy;
    sAf[p * 64 + c] = (f16)v;
  }
  __syncthreads();

  // ---- Phase 2: GEMM1  hdd = lrelu([f_anchor, coords] @ W1 + b1) ----
  // Coord columns (K=64,65) + bias folded into accumulator in VALU, so the
  // WMMA K is exactly 64 = 2 x v_wmma_f32_16x16x32_f16.
  const int n = wave * 16 + ln;    // this lane's HID column
  v8f acc;
  {
    const float w1x = sW1t[0 * 64 + n];
    const float w1y = sW1t[1 * 64 + n];
    const float bb  = sB1[n];
#pragma unroll
    for (int v = 0; v < 8; ++v) {
      const int m = v + mhi;
      acc[v] = bb + sCx[m] * w1x + sCy[m] * w1y;
    }
    v16h a0 = load_A16(sAf, 0);
    v16h b0 = load_Bsw(sW1sw, wave * 2 + 0);
    acc = __builtin_amdgcn_wmma_f32_16x16x32_f16(false, a0, false, b0, (short)0, acc, false, false);
    v16h a1 = load_A16(sAf, 32);
    v16h b1v = load_Bsw(sW1sw, wave * 2 + 1);
    acc = __builtin_amdgcn_wmma_f32_16x16x32_f16(false, a1, false, b1v, (short)0, acc, false, false);
#pragma unroll
    for (int v = 0; v < 8; ++v) acc[v] = acc[v] > 0.0f ? acc[v] : 0.2f * acc[v];
#pragma unroll
    for (int v = 0; v < 8; ++v) sH1[(v + mhi) * 64 + n] = (f16)acc[v];
  }
  __syncthreads();

  // ---- Phase 3: GEMM2  hdd = lrelu(hdd + hdd @ Wr + br) ----
  // Residual add is free: this wave's accumulator already holds the (m,n) tile.
  {
    v8f acc2;
#pragma unroll
    for (int v = 0; v < 8; ++v) acc2[v] = acc[v] + sBr[n];
    v16h a0 = load_A16(sH1, 0);
    v16h b0 = load_Bsw(sWrsw, wave * 2 + 0);
    acc2 = __builtin_amdgcn_wmma_f32_16x16x32_f16(false, a0, false, b0, (short)0, acc2, false, false);
    v16h a1 = load_A16(sH1, 32);
    v16h b1v = load_Bsw(sWrsw, wave * 2 + 1);
    acc2 = __builtin_amdgcn_wmma_f32_16x16x32_f16(false, a1, false, b1v, (short)0, acc2, false, false);
#pragma unroll
    for (int v = 0; v < 8; ++v) acc2[v] = acc2[v] > 0.0f ? acc2[v] : 0.2f * acc2[v];
#pragma unroll
    for (int v = 0; v < 8; ++v) sH2[(v + mhi) * 64 + n] = (f16)acc2[v];
  }
  __syncthreads();

  // ---- Phase 4: GEMM3  routing_out = hdd @ W2 + b2  (27 cols, padded to 32) ----
  if (wave < 2) {                          // wave-uniform: EXEC all-1s in WMMA
    const int n3 = wave * 16 + ln;
    v8f acc3;
#pragma unroll
    for (int v = 0; v < 8; ++v) acc3[v] = sB2[n3];
    v16h a0 = load_A16(sH2, 0);
    v16h b0 = load_Bsw(sW2sw, wave * 2 + 0);
    acc3 = __builtin_amdgcn_wmma_f32_16x16x32_f16(false, a0, false, b0, (short)0, acc3, false, false);
    v16h a1 = load_A16(sH2, 32);
    v16h b1v = load_Bsw(sW2sw, wave * 2 + 1);
    acc3 = __builtin_amdgcn_wmma_f32_16x16x32_f16(false, a1, false, b1v, (short)0, acc3, false, false);
#pragma unroll
    for (int v = 0; v < 8; ++v) sRout[(v + mhi) * 32 + n3] = acc3[v];
  }
  __syncthreads();

  // ---- Phase 5: per-point tanh offsets + softmax weights + tap setup ----
  if (tid < PTS) {
    const float* r = &sRout[tid * 32];
    const float cx = sCx[tid], cy = sCy[tid];
    float mx = r[18];
#pragma unroll
    for (int k = 1; k < K_TAPS; ++k) mx = fmaxf(mx, r[18 + k]);
    float ek[K_TAPS]; float se = 0.0f;
#pragma unroll
    for (int k = 0; k < K_TAPS; ++k) { ek[k] = __expf(r[18 + k] - mx); se += ek[k]; }
    const float inv = 1.0f / se;
#pragma unroll
    for (int k = 0; k < K_TAPS; ++k) {
      const float ox = tanhf(r[2 * k + 0]) * (16.0f / (float)W_DIM);
      const float oy = tanhf(r[2 * k + 1]) * (16.0f / (float)H_DIM);
      int x0, y0, x1, y1; float wx, wy;
      samp_setup(cx + ox, cy + oy, x0, y0, x1, y1, wx, wy);
      const int s = tid * K_TAPS + k;
      sSx0[s] = x0; sSy0[s] = y0; sSx1[s] = x1; sSy1[s] = y1;
      sSwx[s] = wx; sSwy[s] = wy; sSwk[s] = ek[k] * inv;
    }
  }
  __syncthreads();

  // ---- Phase 6: deformable gather + dynamic-weight aggregation ----
  // 8 threads per point, 8 channels per thread.
  {
    const int p  = tid >> 3;
    const int cb = (tid & 7) << 3;
    float accv[8];
#pragma unroll
    for (int c = 0; c < 8; ++c) accv[c] = 0.0f;
    const float* fb = feat + (size_t)sFB[p] + (size_t)cb * (H_DIM * W_DIM);
#pragma unroll
    for (int k = 0; k < K_TAPS; ++k) {
      const int s   = p * K_TAPS + k;
      const int i00 = sSy0[s] * W_DIM + sSx0[s];
      const int i01 = sSy0[s] * W_DIM + sSx1[s];
      const int i10 = sSy1[s] * W_DIM + sSx0[s];
      const int i11 = sSy1[s] * W_DIM + sSx1[s];
      const float wx = sSwx[s], wy = sSwy[s], wk = sSwk[s];
      const float w00 = (1.0f - wx) * (1.0f - wy) * wk;
      const float w01 = wx * (1.0f - wy) * wk;
      const float w10 = (1.0f - wx) * wy * wk;
      const float w11 = wx * wy * wk;
#pragma unroll
      for (int c = 0; c < 8; ++c) {
        const float* f = fb + (size_t)c * (H_DIM * W_DIM);
        accv[c] += w00 * f[i00] + w01 * f[i01] + w10 * f[i10] + w11 * f[i11];
      }
    }
    const int gp = gp0 + p;
    float* o = out + (size_t)gp * C_DIM + cb;
#pragma unroll
    for (int c = 0; c < 8; ++c) o[c] = accv[c];
  }
}

extern "C" void kernel_launch(void* const* d_in, const int* in_sizes, int n_in,
                              void* d_out, int out_size, void* d_ws, size_t ws_size,
                              hipStream_t stream) {
  const float* feat   = (const float*)d_in[0];
  const float* coords = (const float*)d_in[1];
  const float* W1     = (const float*)d_in[2];
  const float* b1     = (const float*)d_in[3];
  const float* Wr     = (const float*)d_in[4];
  const float* br     = (const float*)d_in[5];
  const float* W2     = (const float*)d_in[6];
  const float* b2     = (const float*)d_in[7];
  float* out          = (float*)d_out;

  const int total_pts = in_sizes[1] / 2;          // B * N
  const int grid      = total_pts / PTS;          // 4096 blocks
  deform_dyn_fused_kernel<<<grid, THREADS, 0, stream>>>(
      feat, coords, W1, b1, Wr, br, W2, b2, out);
}
